// BiLSTM_CRF_14826227106703
// MI455X (gfx1250) — compile-verified, matrix operand order checked by reading, not run on hardware
//
#include <hip/hip_runtime.h>

// ---------------- problem constants (match reference) ----------------
#define SEQ   8192
#define EMB   512
#define HHID  512            // per-direction hidden
#define GATES (4*HHID)       // 2048
#define NCAT  (2*GATES)      // 4096 (fwd gates | bwd gates)
#define KTAG  5
#define TSTART 3
#define TSTOP  4
#define NEGV  (-10000.0f)

// ---------------- WMMA vector types (CDNA5 wave32) ----------------
typedef __attribute__((ext_vector_type(16))) __bf16 v16bf;
typedef __attribute__((ext_vector_type(8)))  float  v8f;
typedef __attribute__((ext_vector_type(4)))  unsigned int v4u;
typedef __attribute__((ext_vector_type(4)))  int    v4i;
typedef __attribute__((ext_vector_type(8)))  int    v8i;

union Frag { v16bf v; uint4 q[2]; };   // 32 bytes = 16 bf16

// float -> bf16 (round to nearest even), kept as ushort for portability
static __device__ __forceinline__ unsigned short f2bf(float f) {
    unsigned int u = __float_as_uint(f);
    u += 0x7FFFu + ((u >> 16) & 1u);
    return (unsigned short)(u >> 16);
}

// ---------------- 1) embedding gather + cast to bf16 ----------------
__global__ void bilstm_gather_cast(const int* __restrict__ sent,
                                   const float* __restrict__ embed,
                                   unsigned short* __restrict__ xb) {
    int idx = blockIdx.x * blockDim.x + threadIdx.x;
    if (idx >= SEQ * EMB) return;
    int s = idx / EMB, e = idx - s * EMB;
    xb[idx] = f2bf(embed[(size_t)sent[s] * EMB + e]);
}

// ---------------- 2) pack [W_ih_f ; W_ih_b] as bf16 ----------------
__global__ void bilstm_pack_w(const float* __restrict__ wf,
                              const float* __restrict__ wb,
                              unsigned short* __restrict__ wcat) {
    int idx = blockIdx.x * blockDim.x + threadIdx.x;
    if (idx >= NCAT * EMB) return;
    int row = idx / EMB, col = idx - row * EMB;
    float v = (row < GATES) ? wf[(size_t)row * EMB + col]
                            : wb[(size_t)(row - GATES) * EMB + col];
    wcat[idx] = f2bf(v);
}

// ---------------- 3) input projection GEMM via TDM + WMMA bf16 ----------------
// xp[SEQ][NCAT] = x[SEQ][EMB] @ Wcat^T   (f32 accumulate)
// Block: M-tile 16, N-tile 512 (8 waves x 64N). The 16x512 bf16 A panel is
// DMA'd into LDS once by the Tensor Data Mover; waves feed WMMA A-fragments
// from LDS (ds_load) and B-fragments straight from global.
__global__ void __launch_bounds__(256)
bilstm_inproj_wmma(const unsigned short* __restrict__ xb,
                   const unsigned short* __restrict__ wcat,
                   float* __restrict__ xp) {
    __shared__ __align__(16) unsigned short shA[16 * EMB];   // 16 KB A panel

    const int lane = threadIdx.x & 31;
    const int wave = threadIdx.x >> 5;
    const int m0 = blockIdx.y * 16;
    const int n0 = blockIdx.x * 512 + wave * 64;

#if __has_builtin(__builtin_amdgcn_tensor_load_to_lds) && __has_builtin(__builtin_amdgcn_s_wait_tensorcnt)
    if (wave == 0) {   // TDM is issued per-wave (EXEC ignored): one wave only
        unsigned long long gaddr =
            (unsigned long long)(const void*)(xb + (size_t)m0 * EMB);
        unsigned int ldsoff =
            (unsigned int)(unsigned long long)(void*)&shA[0];  // low 32b = LDS offset
        // ---- D# group 0: count | lds_addr | global_addr | type=2 ----
        v4u g0;
        g0[0] = 1u;                                            // count=1 (valid)
        g0[1] = ldsoff;                                        // lds_addr (bytes)
        g0[2] = (unsigned int)(gaddr & 0xFFFFFFFFull);         // global_addr[31:0]
        g0[3] = (unsigned int)((gaddr >> 32) & 0x1FFFFFFull)   // global_addr[56:32]
              | 0x80000000u;                                   // type=2 ("image")
        // ---- D# group 1: data_size=2B, 2-D tensor/tile dims ----
        v8i g1;
        g1[0] = 0x00010000;                          // data_size=1 -> 2 bytes
        g1[1] = (int)((EMB & 0xFFFF) << 16);         // tensor_dim0[15:0]
        g1[2] = (int)((EMB >> 16) | ((SEQ & 0xFFFF) << 16));   // td0 hi | td1 lo
        g1[3] = (int)((SEQ >> 16) | ((EMB & 0xFFFF) << 16));   // td1 hi | tile_dim0=512
        g1[4] = 16;                                  // tile_dim1=16, tile_dim2=0
        g1[5] = EMB;                                 // tensor_dim0_stride[31:0]
        g1[6] = 0;                                   // stride0 hi | stride1 lo
        g1[7] = 0;                                   // stride1 hi
        v4i z4 = {0, 0, 0, 0};                       // groups 2/3 unused (2-D)
#if __clang_major__ >= 23
        v8i z8 = {0, 0, 0, 0, 0, 0, 0, 0};
        __builtin_amdgcn_tensor_load_to_lds(g0, g1, z4, z4, z8, 0);
#else
        __builtin_amdgcn_tensor_load_to_lds(g0, g1, z4, z4, 0);
#endif
        __builtin_amdgcn_s_wait_tensorcnt(0);
    }
    __syncthreads();
#else
    // fallback: cooperative global->LDS copy
    {
        const uint4* src = (const uint4*)(xb + (size_t)m0 * EMB);
        uint4* dst = (uint4*)shA;
        for (int i = threadIdx.x; i < 16 * EMB / 8; i += blockDim.x)
            dst[i] = src[i];
    }
    __syncthreads();
#endif

    const int mr = lane & 15;        // A row within tile / B column within tile
    const int kh = (lane >> 4) * 8;  // per-lane K sub-offset (ISA bf16 layout)
    const unsigned short* arow = shA + mr * EMB;

    v8f acc[4] = {};
    for (int k0 = 0; k0 < EMB; k0 += 32) {
        Frag a;                                       // from LDS (ds_load_b128)
        a.q[0] = *(const uint4*)(arow + k0 + kh);     // K = k0+kh .. +7
        a.q[1] = *(const uint4*)(arow + k0 + 16 + kh);// K = k0+16+kh .. +7
#pragma unroll
        for (int t = 0; t < 4; ++t) {
            const unsigned short* brow = wcat + (size_t)(n0 + t * 16 + mr) * EMB;
            Frag b;                                   // from global
            b.q[0] = *(const uint4*)(brow + k0 + kh);
            b.q[1] = *(const uint4*)(brow + k0 + 16 + kh);
            acc[t] = __builtin_amdgcn_wmma_f32_16x16x32_bf16(
                false, a.v, false, b.v, (short)0, acc[t], false, false);
        }
    }

    // D layout: lane holds column (lane&15); VGPR r holds row r + 8*(lane>>4)
    const int nn = lane & 15;
    const int mh = (lane >> 4) * 8;
#pragma unroll
    for (int t = 0; t < 4; ++t)
#pragma unroll
        for (int r = 0; r < 8; ++r)
            xp[(size_t)(m0 + mh + r) * NCAT + (n0 + t * 16 + nn)] = acc[t][r];
}

// ---------------- 4) sequential biLSTM recurrence ----------------
// blockIdx.x = direction (0 fwd, 1 bwd). 1024 threads; h lives in LDS.
__global__ void __launch_bounds__(1024)
bilstm_recurrence(const float* __restrict__ xp,
                  const float* __restrict__ Whh_f, const float* __restrict__ b_f,
                  const float* __restrict__ Whh_b, const float* __restrict__ b_b,
                  const float* __restrict__ h0,    const float* __restrict__ c0,
                  float* __restrict__ hf,          float* __restrict__ hb) {
    __shared__ float sh_h[HHID];
    __shared__ float sh_g[GATES];

    const int dir = blockIdx.x;
    const int tid = threadIdx.x;
    const float* Whh  = dir ? Whh_b : Whh_f;
    const float* bias = dir ? b_b   : b_f;
    float*       hs   = dir ? hb    : hf;
    const int    xoff = dir ? GATES : 0;

    float c = 0.0f;
    if (tid < HHID) { sh_h[tid] = h0[dir * HHID + tid]; c = c0[dir * HHID + tid]; }
    __syncthreads();

    const int r0 = tid, r1 = tid + 1024;          // two gate rows per thread
    const float* w0 = Whh + (size_t)r0 * HHID;
    const float* w1 = Whh + (size_t)r1 * HHID;
    const float  bb0 = bias[r0], bb1 = bias[r1];

    for (int n = 0; n < SEQ; ++n) {
        const int t = dir ? (SEQ - 1 - n) : n;
        const float* xr = xp + (size_t)t * NCAT + xoff;
        // hint the next step's precomputed-projection row into cache
        const int tn = dir ? (t - 1) : (t + 1);
        if (tn >= 0 && tn < SEQ)
            __builtin_prefetch(xp + (size_t)tn * NCAT + xoff + r0, 0, 0);

        float a0 = bb0 + xr[r0];
        float a1 = bb1 + xr[r1];
#pragma unroll 4
        for (int k = 0; k < HHID; k += 4) {
            float4 hv  = *(const float4*)&sh_h[k];      // LDS broadcast
            float4 wv0 = *(const float4*)(w0 + k);
            float4 wv1 = *(const float4*)(w1 + k);
            a0 += hv.x * wv0.x + hv.y * wv0.y + hv.z * wv0.z + hv.w * wv0.w;
            a1 += hv.x * wv1.x + hv.y * wv1.y + hv.z * wv1.z + hv.w * wv1.w;
        }
        sh_g[r0] = a0;
        sh_g[r1] = a1;
        __syncthreads();

        if (tid < HHID) {   // gate order i, f, g, o
            float gi = sh_g[tid];
            float gf = sh_g[tid + HHID];
            float gg = sh_g[tid + 2 * HHID];
            float go = sh_g[tid + 3 * HHID];
            float si = 1.0f / (1.0f + __expf(-gi));
            float sf = 1.0f / (1.0f + __expf(-gf));
            float so = 1.0f / (1.0f + __expf(-go));
            c = sf * c + si * tanhf(gg);
            float h = so * tanhf(c);
            sh_h[tid] = h;
            hs[(size_t)t * HHID + tid] = h;
        }
        __syncthreads();
    }
}

// ---------------- 5) feats = [hf|hb] @ W_out^T + b_out ----------------
__global__ void bilstm_feats(const float* __restrict__ hf,
                             const float* __restrict__ hb,
                             const float* __restrict__ Wout,
                             const float* __restrict__ bout,
                             float* __restrict__ feats) {
    int idx = blockIdx.x * blockDim.x + threadIdx.x;
    if (idx >= SEQ * KTAG) return;
    int s = idx / KTAG, k = idx - s * KTAG;
    const float* w   = Wout + (size_t)k * (2 * HHID);
    const float* hfs = hf + (size_t)s * HHID;
    const float* hbs = hb + (size_t)s * HHID;
    float acc = bout[k];
    for (int j = 0; j < HHID; ++j) acc += w[j] * hfs[j];
    for (int j = 0; j < HHID; ++j) acc += w[HHID + j] * hbs[j];
    feats[idx] = acc;
}

// ---------------- 6) Viterbi decode: one wave32, lane = next tag ----------------
__global__ void bilstm_viterbi(const float* __restrict__ feats,
                               const float* __restrict__ trans,
                               int* __restrict__ bptr,
                               float* __restrict__ out) {
    const int lane = threadIdx.x;               // 32 lanes, lanes 0..4 active tags
    float fv = (lane == TSTART) ? 0.0f : NEGV;
    float tr[KTAG];
#pragma unroll
    for (int p = 0; p < KTAG; ++p)
        tr[p] = (lane < KTAG) ? trans[lane * KTAG + p] : NEGV;

    for (int s = 0; s < SEQ; ++s) {
        float best = -3.0e38f; int arg = 0;
#pragma unroll
        for (int p = 0; p < KTAG; ++p) {
            float sc = __shfl(fv, p, 32) + tr[p];
            if (sc > best) { best = sc; arg = p; }
        }
        float feat = (lane < KTAG) ? feats[s * KTAG + lane] : 0.0f;
        fv = best + feat;
        if (lane < KTAG) bptr[s * KTAG + lane] = arg;
    }

    float term = fv + ((lane < KTAG) ? trans[TSTOP * KTAG + lane] : NEGV);
    float bestv = -3.0e38f; int besti = 0;
#pragma unroll
    for (int p = 0; p < KTAG; ++p) {
        float v = __shfl(term, p, 32);
        if (v > bestv) { bestv = v; besti = p; }
    }
    if (lane == 0) {
        out[0] = bestv;
        int tag = besti;
        for (int s = SEQ - 1; s >= 0; --s) {
            out[1 + s] = (float)tag;
            tag = bptr[s * KTAG + tag];
        }
    }
}

// ---------------- launcher ----------------
extern "C" void kernel_launch(void* const* d_in, const int* in_sizes, int n_in,
                              void* d_out, int out_size, void* d_ws, size_t ws_size,
                              hipStream_t stream) {
    const int*   sentence = (const int*)  d_in[0];
    const float* embed    = (const float*)d_in[1];
    const float* W_ih_f   = (const float*)d_in[2];
    const float* W_hh_f   = (const float*)d_in[3];
    const float* b_f      = (const float*)d_in[4];
    const float* W_ih_b   = (const float*)d_in[5];
    const float* W_hh_b   = (const float*)d_in[6];
    const float* b_b      = (const float*)d_in[7];
    const float* W_out    = (const float*)d_in[8];
    const float* b_out    = (const float*)d_in[9];
    const float* trans    = (const float*)d_in[10];
    const float* h0       = (const float*)d_in[11];
    const float* c0       = (const float*)d_in[12];
    float* out = (float*)d_out;

    char* ws = (char*)d_ws;
    unsigned short* xb   = (unsigned short*)ws; ws += (size_t)SEQ * EMB * 2;   //   8 MB
    unsigned short* wcat = (unsigned short*)ws; ws += (size_t)NCAT * EMB * 2;  //   4 MB
    float* xp    = (float*)ws; ws += (size_t)SEQ * NCAT * 4;                   // 128 MB
    float* hf    = (float*)ws; ws += (size_t)SEQ * HHID * 4;                   //  16 MB
    float* hb    = (float*)ws; ws += (size_t)SEQ * HHID * 4;                   //  16 MB
    float* feats = (float*)ws; ws += (size_t)SEQ * KTAG * 4;
    int*   bptr  = (int*)ws;

    bilstm_gather_cast<<<(SEQ * EMB + 255) / 256, 256, 0, stream>>>(sentence, embed, xb);
    bilstm_pack_w<<<(NCAT * EMB + 255) / 256, 256, 0, stream>>>(W_ih_f, W_ih_b, wcat);

    dim3 gg(NCAT / 512, SEQ / 16);        // 8 x 512 blocks, 8 waves/block
    bilstm_inproj_wmma<<<gg, 256, 0, stream>>>(xb, wcat, xp);

    bilstm_recurrence<<<2, 1024, 0, stream>>>(xp, W_hh_f, b_f, W_hh_b, b_b,
                                              h0, c0, hf, hb);

    bilstm_feats<<<(SEQ * KTAG + 255) / 256, 256, 0, stream>>>(hf, hb, W_out, b_out, feats);
    bilstm_viterbi<<<1, 32, 0, stream>>>(feats, trans, bptr, out);
}
